// SRSelfAttention_54451595378950
// MI455X (gfx1250) — matrix operand order repaired and checked by reading, not compiled, not run on hardware
//
#include <hip/hip_runtime.h>

// ---------------------------------------------------------------------------
// SR self-attention for MI455X (gfx1250, wave32, WMMA bf16 16x16x32)
// Data movement: TDM-resident B panels + async double-buffered A tiles.
// ---------------------------------------------------------------------------

typedef unsigned short bh;                                   // bf16 bit pattern
typedef float v8f  __attribute__((ext_vector_type(8)));
typedef __bf16 v16bf __attribute__((ext_vector_type(16)));
typedef unsigned int v4u __attribute__((ext_vector_type(4)));
typedef int v8i __attribute__((ext_vector_type(8)));
typedef int v4i __attribute__((ext_vector_type(4)));

union FragU { uint4 u[2]; v16bf v; };                        // 32B A/B fragment

#define DIM   256
#define HEADS 8
#define HD    32
#define BATCH 8
#define NTOK  3137     // 1 + 56*56
#define MSR   785      // 1 + 28*28
#define MPAD  800      // padded KV length (multiple of 32)
#define NPIX  784      // 28*28
#define MQ    (BATCH * NTOK)   // 25096
#define MKV   (BATCH * MSR)    // 6280
#define MCONV (BATCH * NPIX)   // 6272

__device__ __forceinline__ bh f2b(float f) {
  unsigned u = __float_as_uint(f);
  unsigned r = u + 0x7FFFu + ((u >> 16) & 1u);               // round-nearest-even
  return (bh)(r >> 16);
}

__device__ __forceinline__ v8f vzero8() {
  v8f z;
#pragma unroll
  for (int i = 0; i < 8; ++i) z[i] = 0.0f;
  return z;
}

// Async global -> LDS copy (128b), ASYNCcnt-tracked, bypasses VGPRs.
__device__ __forceinline__ void async_b128(unsigned lds_off, const void* gptr) {
  asm volatile("global_load_async_to_lds_b128 %0, %1, off"
               :: "v"(lds_off), "v"((unsigned long long)(uintptr_t)gptr)
               : "memory");
}
__device__ __forceinline__ void wait_async0() {
  asm volatile("s_wait_asynccnt 0x0" ::: "memory");
}

// ---------------------------------------------------------------------------
// Pack kernels
// ---------------------------------------------------------------------------
__global__ __launch_bounds__(256) void f2b_kernel(const float* __restrict__ in,
                                                  bh* __restrict__ out, int n) {
  for (int i = blockIdx.x * blockDim.x + threadIdx.x; i < n;
       i += gridDim.x * blockDim.x)
    out[i] = f2b(in[i]);
}

// conv_w [co,ci,kh,kw] -> W2 [(kh*2+kw)*256+ci , co]  (im2col weight, bf16)
__global__ __launch_bounds__(256) void convw_pack_kernel(const float* __restrict__ w,
                                                         bh* __restrict__ out) {
  int i = blockIdx.x * blockDim.x + threadIdx.x;             // over 1024*256
  if (i >= 1024 * 256) return;
  int co = i & 255;
  int kk = i >> 8;
  int ci = kk & 255;
  int seg = kk >> 8;                                         // kh*2+kw
  int kh = seg >> 1, kw = seg & 1;
  out[i] = f2b(w[((size_t)(co * 256 + ci) * 2 + kh) * 2 + kw]);
}

// cls token row: xr[b,0,:] = bf16(x[b,0,:])
__global__ __launch_bounds__(256) void cls_pack_kernel(const float* __restrict__ x,
                                                       bh* __restrict__ xrh) {
  int i = blockIdx.x * blockDim.x + threadIdx.x;             // over 8*256
  if (i >= BATCH * DIM) return;
  int b = i >> 8, c = i & 255;
  xrh[(size_t)(b * MSR) * DIM + c] = f2b(x[(size_t)b * NTOK * DIM + c]);
}

// ---------------------------------------------------------------------------
// bf16 WMMA GEMM: C[M,N] = A[M,K] * B[K,N].  Block tile 128x64, 8 waves,
// each wave 32x32 (2x2 WMMA tiles).  A tiles: async-to-LDS, double buffered.
// BPANEL (K==256): whole B panel (256 x 64) DMA'd once by the TDM with LDS
// row padding to 72 elements (144B); no per-tile B staging.
// MODE 0: Q (bf16, *scale)  1: conv (f32, im2col A gather)
// MODE 2: KV (K stored transposed [b,h,32,MPAD], V row-major bf16)
// MODE 3: proj (f32 + bias)
// ---------------------------------------------------------------------------
template <int MODE, bool CONVA, bool BPANEL>
__global__ __launch_bounds__(256) void gemm_wmma(
    const bh* __restrict__ A, const bh* __restrict__ Bm,
    int M, int K, int N,
    bh* __restrict__ outb, float* __restrict__ outf,
    const float* __restrict__ bias,
    bh* __restrict__ kT, bh* __restrict__ vrows, float scale) {
  __shared__ __align__(16) bh As[2][128][40];                // padded stride
  __shared__ __align__(16) bh Bst[BPANEL ? (256 * 72) : (2 * 32 * 72)];

  const int tid  = threadIdx.x;
  const int wave = tid >> 5, lane = tid & 31;
  const int wm = wave & 3, wn = wave >> 2;                   // 4x2 wave grid
  const int lhalf = lane >> 4, lmod = lane & 15;
  const int m0 = blockIdx.y * 128;
  const int n0 = blockIdx.x * 64;

  const int ar = tid >> 1, ah = tid & 1;                     // A loader: 2 thr/row
  const int br = tid >> 3, bc = (tid & 7) * 8;               // B loader: 8 thr/row

  v8f acc[2][2];
#pragma unroll
  for (int i = 0; i < 2; ++i)
#pragma unroll
    for (int j = 0; j < 2; ++j) acc[i][j] = vzero8();

  // ---- async stage issue for K-chunk kk into buffer buf ----
  auto issueA = [&](int kk, int buf) {
    const bh* srcA;
    int gr = m0 + ar;
    if constexpr (CONVA) {
      int bb = gr / NPIX, p = gr % NPIX;
      int oh = p / 28, ow = p % 28;
      int seg = kk >> 8, kh = seg >> 1, kw = seg & 1;
      int tok = 1 + (2 * oh + kh) * 56 + (2 * ow + kw);
      int off = (kk & 255) + ah * 16;
      srcA = A + ((size_t)(bb * NTOK + tok) * DIM + off);
    } else {
      int grc = gr < M ? gr : M - 1;                         // clamp: padded rows
      srcA = A + ((size_t)grc * K + kk + ah * 16);           // feed discarded C rows
    }
    unsigned l0 = (unsigned)(uintptr_t)&As[buf][ar][ah * 16];
    async_b128(l0, srcA);
    async_b128(l0 + 16, srcA + 8);
  };
  auto issueB = [&](int kk, int buf) {                       // conv path only
    const bh* srcB = Bm + ((size_t)(kk + br) * N + n0 + bc);
    unsigned l0 = (unsigned)(uintptr_t)&Bst[(size_t)(buf * 32 + br) * 72 + bc];
    async_b128(l0, srcB);
  };

  // ---- TDM: DMA whole B panel [256 x 64] once (BPANEL mode) ----
  if constexpr (BPANEL) {
    if (wave == 0) {
      unsigned long long ga = (unsigned long long)(uintptr_t)(Bm + n0);
      unsigned ldsb = (unsigned)(uintptr_t)&Bst[0];
      v4u g0;
      g0[0] = 1u;                                            // count=1
      g0[1] = ldsb;                                          // lds_addr
      g0[2] = (unsigned)(ga & 0xFFFFFFFFu);                  // global_addr lo
      g0[3] = (unsigned)((ga >> 32) & 0x01FFFFFFu) | (2u << 30);  // hi | type=2
      v8i g1;
      g1[0] = (int)((1u << 16) |                             // data_size = 2B
                    (1u << 20) |                             // pad_enable
                    (4u << 22) |                             // pad every 32 DW (128B)
                    (3u << 25));                             // pad 4 DW (16B)
      g1[1] = (int)((unsigned)(N & 0xFFFF) << 16);           // tensor_dim0 lo16
      g1[2] = (int)(((unsigned)N >> 16) | (256u << 16));     // dim0 hi | dim1 lo
      g1[3] = (int)(64u << 16);                              // dim1 hi=0 | tile_dim0=64
      g1[4] = (int)256u;                                     // tile_dim1=256, tile_dim2=0
      g1[5] = (int)N;                                        // dim0_stride lo32
      g1[6] = 0;                                             // stride hi / dim1_stride
      g1[7] = 0;
      v4i z4;
      for (int i = 0; i < 4; ++i) z4[i] = 0;
#if defined(__clang_major__) && __clang_major__ >= 23
      v8i z8;
      for (int i = 0; i < 8; ++i) z8[i] = 0;
      __builtin_amdgcn_tensor_load_to_lds(g0, g1, z4, z4, z8, 0);
#else
      __builtin_amdgcn_tensor_load_to_lds(g0, g1, z4, z4, 0);
#endif
    }
  }

  issueA(0, 0);
  if constexpr (!BPANEL) issueB(0, 0);
  if constexpr (BPANEL) {
    if (wave == 0) __builtin_amdgcn_s_wait_tensorcnt((short)0);
  }

  const int nkt = K >> 5;
  for (int kt = 0; kt < nkt; ++kt) {
    wait_async0();                                           // this tile landed
    __syncthreads();                                         // all waves' tiles
    if (kt + 1 < nkt) {                                      // overlap next stage
      issueA((kt + 1) << 5, (kt + 1) & 1);
      if constexpr (!BPANEL) issueB((kt + 1) << 5, (kt + 1) & 1);
    }

    // ---- fragments + WMMA ----
    const int buf = kt & 1;
    FragU afr[2], bfr[2];
#pragma unroll
    for (int i = 0; i < 2; ++i) {
      int row = wm * 32 + i * 16 + lmod;
      int kb = lhalf * 8;                                    // A lane layout
      afr[i].u[0] = *(const uint4*)&As[buf][row][kb];
      afr[i].u[1] = *(const uint4*)&As[buf][row][kb + 16];
    }
#pragma unroll
    for (int j = 0; j < 2; ++j) {
      int col = wn * 32 + j * 16 + lmod;                     // B: lane = N col, elems = K (table order)
      int krow = BPANEL ? (kt << 5) : (buf * 32);
#pragma unroll
      for (int s = 0; s < 16; ++s) {
        int k = (s < 8) ? (lhalf * 8 + s) : (16 + lhalf * 8 + s - 8);
        bfr[j].v[s] = __builtin_bit_cast(__bf16, Bst[(size_t)(krow + k) * 72 + col]);
      }
    }
#pragma unroll
    for (int i = 0; i < 2; ++i)
#pragma unroll
      for (int j = 0; j < 2; ++j)
        acc[i][j] = __builtin_amdgcn_wmma_f32_16x16x32_bf16(
            false, afr[i].v, false, bfr[j].v, (short)0, acc[i][j], false, false);
  }

  // ---- epilogue: C layout M=g+8*lhalf, N=lmod ----
#pragma unroll
  for (int i = 0; i < 2; ++i)
#pragma unroll
    for (int j = 0; j < 2; ++j)
#pragma unroll
      for (int g = 0; g < 8; ++g) {
        int R = m0 + wm * 32 + i * 16 + g + lhalf * 8;
        int C = n0 + wn * 32 + j * 16 + lmod;
        if (R >= M) continue;
        float v = acc[i][j][g];
        if constexpr (MODE == 0) {                           // Q (scaled bf16)
          outb[(size_t)R * N + C] = f2b(v * scale);
        } else if constexpr (MODE == 1) {                    // conv f32
          outf[(size_t)R * N + C] = v;
        } else if constexpr (MODE == 2) {                    // KV split
          int bb = R / MSR, mi = R % MSR;
          if (C < DIM) {                                     // K -> transposed
            int hh = C >> 5, dd = C & 31;
            kT[((size_t)(bb * HEADS + hh) * HD + dd) * MPAD + mi] = f2b(v);
          } else {                                           // V row-major
            vrows[((size_t)(bb * MPAD) + mi) * DIM + (C - DIM)] = f2b(v);
          }
        } else {                                             // proj + bias f32
          outf[(size_t)R * N + C] = v + bias[C];
        }
      }
}

// ---------------------------------------------------------------------------
// Instance norm over 784 pixels per (b, channel); writes xr rows 1..784 (bf16)
// ---------------------------------------------------------------------------
__global__ __launch_bounds__(128) void inorm_kernel(const float* __restrict__ conv,
                                                    bh* __restrict__ xrh) {
  const int co = blockIdx.x, b = blockIdx.y, t = threadIdx.x;
  __shared__ float s1[128], s2[128];
  float a = 0.f, q = 0.f;
  for (int p = t; p < NPIX; p += 128) {
    float v = conv[((size_t)(b * NPIX) + p) * DIM + co];
    a += v; q += v * v;
  }
  s1[t] = a; s2[t] = q;
  __syncthreads();
  for (int s = 64; s > 0; s >>= 1) {
    if (t < s) { s1[t] += s1[t + s]; s2[t] += s2[t + s]; }
    __syncthreads();
  }
  float mean = s1[0] * (1.0f / NPIX);
  float var  = s2[0] * (1.0f / NPIX) - mean * mean;
  float inv  = rsqrtf(var + 1e-5f);
  for (int p = t; p < NPIX; p += 128) {
    float v = conv[((size_t)(b * NPIX) + p) * DIM + co];
    xrh[((size_t)(b * MSR) + 1 + p) * DIM + co] = f2b((v - mean) * inv);
  }
}

// ---------------------------------------------------------------------------
// Attention: block = (16 queries, head, batch). 8 waves.
// Phase 1: S = Q K^T via WMMA (K pre-transposed -> contiguous B frags)
// Phase 2: block softmax (unnormalized P in bf16 LDS, masked >= 785)
// Phase 3: O = P V via WMMA, cross-wave reduce, normalize at store.
// ---------------------------------------------------------------------------
__global__ __launch_bounds__(256) void attn_kernel(
    const bh* __restrict__ qh,   // [b,3137,256] (already * 1/sqrt(dh))
    const bh* __restrict__ kT,   // [b,8,32,MPAD]
    const bh* __restrict__ vh,   // [b,MPAD,256]
    bh* __restrict__ obf) {      // [b,3137,256]
  const int n0 = blockIdx.x * 16;
  const int h  = blockIdx.y;
  const int b  = blockIdx.z;
  const int tid = threadIdx.x;
  const int wave = tid >> 5, lane = tid & 31;
  const int lhalf = lane >> 4, lmod = lane & 15;

  __shared__ __align__(16) float Sld[16][MPAD];              // 51200 B
  __shared__ __align__(16) bh    Pld[16][MPAD];              // 25600 B
  __shared__ __align__(16) float Opart[8][16][HD];           // 16384 B
  __shared__ float red[16][16], rowmax[16], rowsum[16];

  // Q fragment (loop invariant). A layout: M=lmod, lane half h holds K 8h..8h+7, 16+8h..16+8h+7 (table order).
  int qrow = n0 + lmod; if (qrow > NTOK - 1) qrow = NTOK - 1;
  const bh* qp = qh + ((size_t)(b * NTOK + qrow) * DIM + h * HD + lhalf * 8);
  FragU qf;
  qf.u[0] = *(const uint4*)qp;
  qf.u[1] = *(const uint4*)(qp + 16);

  // ---- Phase 1: S tiles over 50 key tiles of 16 ----
  for (int kt = wave; kt * 16 < MPAD; kt += 8) {
    const bh* kp = kT + ((size_t)(b * HEADS + h) * HD * MPAD + kt * 16 + lmod);
    FragU kf;
#pragma unroll
    for (int s = 0; s < 16; ++s) {
      int d = (s < 8) ? (lhalf * 8 + s) : (16 + lhalf * 8 + s - 8);
      kf.v[s] = __builtin_bit_cast(__bf16, kp[(size_t)d * MPAD]);
    }
    v8f s = vzero8();
    s = __builtin_amdgcn_wmma_f32_16x16x32_bf16(false, qf.v, false, kf.v,
                                                (short)0, s, false, false);
#pragma unroll
    for (int g = 0; g < 8; ++g) {
      int r = g + lhalf * 8;
      int j = kt * 16 + lmod;
      Sld[r][j] = (j < MSR) ? s[g] : -1e30f;
    }
  }
  __syncthreads();

  // ---- Phase 2: softmax (16 threads per query row) ----
  {
    const int r = tid >> 4, c = tid & 15;
    float mx = -1e30f;
    for (int j = c; j < MSR; j += 16) mx = fmaxf(mx, Sld[r][j]);
    red[r][c] = mx;
    __syncthreads();
    if (c == 0) {
      float m2 = -1e30f;
      for (int t = 0; t < 16; ++t) m2 = fmaxf(m2, red[r][t]);
      rowmax[r] = m2;
    }
    __syncthreads();
    float mrow = rowmax[r];
    float sm = 0.f;
    for (int j = c; j < MPAD; j += 16) {
      if (j < MSR) {
        float e = __expf(Sld[r][j] - mrow);
        sm += e;
        Pld[r][j] = f2b(e);
      } else {
        Pld[r][j] = 0;
      }
    }
    red[r][c] = sm;
    __syncthreads();
    if (c == 0) {
      float s2 = 0.f;
      for (int t = 0; t < 16; ++t) s2 += red[r][t];
      rowsum[r] = s2;
    }
    __syncthreads();
  }

  // ---- Phase 3: O = P V over 25 key chunks of 32 ----
  v8f o0 = vzero8(), o1 = vzero8();
  for (int ck = wave; ck * 32 < MPAD; ck += 8) {
    FragU pf;                                                // A: P 16x32
    int kb = ck * 32 + lhalf * 8;
    pf.u[0] = *(const uint4*)&Pld[lmod][kb];
    pf.u[1] = *(const uint4*)&Pld[lmod][kb + 16];
    const bh* vp = vh + ((size_t)(b * MPAD + ck * 32) * DIM + h * HD + lmod);  // B: lane = d col, elems = key
    FragU vf0, vf1;
#pragma unroll
    for (int s = 0; s < 16; ++s) {
      int key = (s < 8) ? (lhalf * 8 + s) : (16 + lhalf * 8 + s - 8);
      vf0.v[s] = __builtin_bit_cast(__bf16, vp[(size_t)key * DIM]);
      vf1.v[s] = __builtin_bit_cast(__bf16, vp[(size_t)key * DIM + 16]);
    }
    o0 = __builtin_amdgcn_wmma_f32_16x16x32_bf16(false, pf.v, false, vf0.v,
                                                 (short)0, o0, false, false);
    o1 = __builtin_amdgcn_wmma_f32_16x16x32_bf16(false, pf.v, false, vf1.v,
                                                 (short)0, o1, false, false);
  }
#pragma unroll
  for (int g = 0; g < 8; ++g) {
    int qr = g + lhalf * 8;
    Opart[wave][qr][lmod]      = o0[g];
    Opart[wave][qr][lmod + 16] = o1[g];
  }
  __syncthreads();

  for (int idx = tid; idx < 16 * HD; idx += 256) {
    int qr = idx >> 5, dd = idx & 31;
    float s = 0.f;
#pragma unroll
    for (int w = 0; w < 8; ++w) s += Opart[w][qr][dd];
    s /= rowsum[qr];
    int n = n0 + qr;
    if (n < NTOK)
      obf[((size_t)(b * NTOK) + n) * DIM + h * HD + dd] = f2b(s);
  }
}

// ---------------------------------------------------------------------------
// Host-side launch
// ---------------------------------------------------------------------------
static inline size_t au(size_t x) { return (x + 255) & ~(size_t)255; }

extern "C" void kernel_launch(void* const* d_in, const int* in_sizes, int n_in,
                              void* d_out, int out_size, void* d_ws, size_t ws_size,
                              hipStream_t stream) {
  const float* x      = (const float*)d_in[0];
  const float* Wq     = (const float*)d_in[1];
  const float* Wkv    = (const float*)d_in[2];
  const float* conv_w = (const float*)d_in[3];
  const float* Wproj  = (const float*)d_in[4];
  const float* bproj  = (const float*)d_in[5];
  float* out = (float*)d_out;

  char* ws = (char*)d_ws;
  size_t off = 0;
  bh* xh    = (bh*)(ws + off); off = au(off + (size_t)MQ * DIM * 2);
  bh* qh    = (bh*)(ws + off); off = au(off + (size_t)MQ * DIM * 2);
  bh* wqh   = (bh*)(ws + off); off = au(off + (size_t)DIM * DIM * 2);
  bh* wkvh  = (bh*)(ws + off); off = au(off + (size_t)DIM * 2 * DIM * 2);
  bh* wprojh= (bh*)(ws + off); off = au(off + (size_t)DIM * DIM * 2);
  bh* w2h   = (bh*)(ws + off); off = au(off + (size_t)1024 * DIM * 2);
  float* convf = (float*)(ws + off); off = au(off + (size_t)MCONV * DIM * 4);
  bh* xrh   = (bh*)(ws + off); off = au(off + (size_t)MKV * DIM * 2);
  bh* kTh   = (bh*)(ws + off); off = au(off + (size_t)BATCH * HEADS * HD * MPAD * 2);
  bh* vhh   = (bh*)(ws + off); off = au(off + (size_t)BATCH * MPAD * DIM * 2);
  bh* obf   = (bh*)(ws + off); off = au(off + (size_t)MQ * DIM * 2);
  (void)ws_size; (void)n_in; (void)in_sizes; (void)out_size;

  const float qscale = 0.17677669529663687f;                 // 1/sqrt(32)

  // 1) pack to bf16
  f2b_kernel<<<dim3(4096), 256, 0, stream>>>(x, xh, MQ * DIM);
  f2b_kernel<<<dim3(256), 256, 0, stream>>>(Wq, wqh, DIM * DIM);
  f2b_kernel<<<dim3(512), 256, 0, stream>>>(Wkv, wkvh, DIM * 2 * DIM);
  f2b_kernel<<<dim3(256), 256, 0, stream>>>(Wproj, wprojh, DIM * DIM);
  convw_pack_kernel<<<dim3(1024), 256, 0, stream>>>(conv_w, w2h);
  cls_pack_kernel<<<dim3(8), 256, 0, stream>>>(x, xrh);

  // 2) Q = (x @ Wq) * scale  -> bf16      (TDM B panel)
  gemm_wmma<0, false, true><<<dim3(4, 197), 256, 0, stream>>>(
      xh, wqh, MQ, DIM, DIM, qh, nullptr, nullptr, nullptr, nullptr, qscale);

  // 3) conv (im2col GEMM, K=1024) -> f32  (async-staged A and B)
  gemm_wmma<1, true, false><<<dim3(4, 49), 256, 0, stream>>>(
      xh, w2h, MCONV, 1024, DIM, nullptr, convf, nullptr, nullptr, nullptr, 1.f);

  // 4) instance norm -> xr rows 1..784
  inorm_kernel<<<dim3(DIM, BATCH), 128, 0, stream>>>(convf, xrh);

  // 5) KV = xr @ Wkv  (K transposed per head, V row-major; TDM B panel)
  gemm_wmma<2, false, true><<<dim3(8, 50), 256, 0, stream>>>(
      xrh, wkvh, MKV, DIM, 2 * DIM, nullptr, nullptr, nullptr, kTh, vhh, 1.f);

  // 6) attention
  attn_kernel<<<dim3(197, HEADS, BATCH), 256, 0, stream>>>(qh, kTh, vhh, obf);

  // 7) projection + bias -> f32 output   (TDM B panel)
  gemm_wmma<3, false, true><<<dim3(4, 197), 256, 0, stream>>>(
      obf, wprojh, MQ, DIM, DIM, nullptr, out, bproj, nullptr, nullptr, 1.f);
}
